// STBlocks_27556510171432
// MI455X (gfx1250) — compile-verified, matrix operand order checked by reading, not run on hardware
//
#include <hip/hip_runtime.h>
#include <math.h>

// ---------------------------------------------------------------------------
// STBlocks (hypergraph GRU) for MI455X / gfx1250.
// Dense per-step GEMMs run on v_wmma_f32_16x16x32_f16 (wave32 WMMA) with
// operands pre-swizzled into WMMA fragment layout so the inner loop is just
// two 32B vector loads + one WMMA per k-chunk.
// Hypergraph conv = atomic scatter/gather (bandwidth bound, L2-resident eagg).
// ---------------------------------------------------------------------------

typedef _Float16 v16h __attribute__((ext_vector_type(16)));
typedef float    v8f  __attribute__((ext_vector_type(8)));

constexpr int B_    = 4;
constexpr int T_    = 12;
constexpr int C_    = 64;
constexpr int N_    = 10000;
constexpr int H_    = 64;
constexpr int L_    = 2;
constexpr int NE_   = 2000;
constexpr int NNZ_  = 80000;
constexpr int TOPK_ = 10;
constexpr int DE_   = 16;
constexpr int D_    = C_ + H_;      // 128
constexpr float RATE_ = 0.5f;

// Fragment packing helper (device+host agree on layout):
// 16-bit WMMA A/B fragment for a 16(x/)32 tile: lane = (kh<<4)|idx16,
// element j<8 -> k = kh*8+j ; j>=8 -> k = 16+kh*8+(j-8).
// Packed tensor layout: [tile_m or (ck,tn)][lane][j], 16 halves per lane.

// ---------------------------------------------------------------- utilities
__global__ void zero_f32(float* __restrict__ p, long long n) {
    long long i = (long long)blockIdx.x * blockDim.x + threadIdx.x;
    if (i < n) p[i] = 0.0f;
}

// Pack W (KxNout, f32, row-major) into WMMA B-fragment order, f16.
// dst index i = ((ck*nTilesN + tn)*32 + lane)*16 + j
__global__ void pack_weight(const float* __restrict__ W, _Float16* __restrict__ P,
                            int K, int Nout) {
    int i = blockIdx.x * blockDim.x + threadIdx.x;
    if (i >= K * Nout) return;
    int j    = i & 15;
    int lane = (i >> 4) & 31;
    int tl   = i >> 9;
    int nTilesN = Nout >> 4;
    int tn = tl % nTilesN;
    int ck = tl / nTilesN;
    int n  = tn * 16 + (lane & 15);
    int kh = lane >> 4;
    int k  = ck * 32 + ((j < 8) ? (kh * 8 + j) : (16 + kh * 8 + (j - 8)));
    P[i] = (_Float16)W[(size_t)k * Nout + n];
}

// ------------------------------------------------------------------ degrees
__global__ void count_deg(const int* __restrict__ node_idx, const int* __restrict__ edge_idx,
                          float* __restrict__ deg_n, float* __restrict__ deg_e) {
    int i = blockIdx.x * blockDim.x + threadIdx.x;
    if (i < NNZ_) {
        atomicAdd(&deg_e[edge_idx[i]], 1.0f);
        atomicAdd(&deg_n[node_idx[i]], 1.0f);
    }
}

__global__ void invert_deg(float* __restrict__ p, int n) {
    int i = blockIdx.x * blockDim.x + threadIdx.x;
    if (i < n) p[i] = 1.0f / fmaxf(p[i], 1.0f);
}

// -------------------------------------------------------- adaptive top-k/row
__global__ void topk_kernel(const float* __restrict__ E,
                            int* __restrict__ tidx, float* __restrict__ tw) {
    __shared__ float sv[128 * TOPK_];
    __shared__ int   si[128 * TOPK_];
    const int n   = blockIdx.x;
    const int tid = threadIdx.x;

    float en[DE_];
#pragma unroll
    for (int d = 0; d < DE_; ++d) en[d] = E[(size_t)n * DE_ + d];

    float bv[TOPK_];
    int   bi[TOPK_];
#pragma unroll
    for (int k = 0; k < TOPK_; ++k) { bv[k] = -1.0f; bi[k] = 0; }

    for (int m = tid; m < N_; m += 128) {
        float s = 0.0f;
#pragma unroll
        for (int d = 0; d < DE_; ++d) s += en[d] * E[(size_t)m * DE_ + d];
        s = fmaxf(s, 0.0f);                               // relu
        if (s > bv[TOPK_ - 1]) {
            int p = TOPK_ - 1;
            while (p > 0 && bv[p - 1] < s) {
                bv[p] = bv[p - 1]; bi[p] = bi[p - 1]; --p;
            }
            bv[p] = s; bi[p] = m;
        }
    }
#pragma unroll
    for (int k = 0; k < TOPK_; ++k) {
        sv[tid * TOPK_ + k] = bv[k];
        si[tid * TOPK_ + k] = bi[k];
    }
    __syncthreads();

    if (tid == 0) {
        float vals[TOPK_]; int idxs[TOPK_];
        for (int k = 0; k < TOPK_; ++k) {
            float best = -2.0f; int bestj = 0;
            for (int j = 0; j < 128 * TOPK_; ++j) {
                if (sv[j] > best) { best = sv[j]; bestj = j; }
            }
            vals[k] = best; idxs[k] = si[bestj];
            sv[bestj] = -2.0f;
        }
        float mx = vals[0], sum = 0.0f;
        float e[TOPK_];
        for (int k = 0; k < TOPK_; ++k) { e[k] = __expf(vals[k] - mx); sum += e[k]; }
        float inv = 1.0f / sum;
        for (int k = 0; k < TOPK_; ++k) {
            tidx[(size_t)n * TOPK_ + k] = idxs[k];
            tw  [(size_t)n * TOPK_ + k] = e[k] * inv;
        }
    }
}

// ----------------------------------------------------------------- build f
__global__ void build_f(const float* __restrict__ xsrc, const float* __restrict__ h,
                        const float* __restrict__ zr, float* __restrict__ f,
                        int t, int mode) {
    long long i = (long long)blockIdx.x * blockDim.x + threadIdx.x;
    if (i >= (long long)B_ * N_ * D_) return;
    int d = (int)(i % D_);
    long long bn = i / D_;
    int n = (int)(bn % N_);
    int b = (int)(bn / N_);
    float v;
    if (d < C_) {
        v = xsrc[(((size_t)b * T_ + t) * C_ + d) * N_ + n];
    } else {
        float hv = h[bn * H_ + (d - C_)];
        if (mode) hv *= zr[bn * (2 * H_) + H_ + (d - C_)];
        v = hv;
    }
    f[i] = v;
}

// ------------------------------------------------------------- conv scatter
__global__ void scatter_edges(const float* __restrict__ f,
                              const int* __restrict__ node_idx,
                              const int* __restrict__ edge_idx,
                              float* __restrict__ eagg) {
    int nz = blockIdx.x % NNZ_;
    int b  = blockIdx.x / NNZ_;
    int d  = threadIdx.x;                       // 128
    int node = node_idx[nz];
    int e    = edge_idx[nz];
    atomicAdd(&eagg[((size_t)b * NE_ + e) * D_ + d],
              f[((size_t)b * N_ + node) * D_ + d]);
}

__global__ void scatter_nodes(const float* __restrict__ eagg,
                              const int* __restrict__ node_idx,
                              const int* __restrict__ edge_idx,
                              const float* __restrict__ inv_e,
                              float* __restrict__ gstat) {
    int nz = blockIdx.x % NNZ_;
    int b  = blockIdx.x / NNZ_;
    int d  = threadIdx.x;
    int node = node_idx[nz];
    int e    = edge_idx[nz];
    atomicAdd(&gstat[((size_t)b * N_ + node) * D_ + d],
              inv_e[e] * eagg[((size_t)b * NE_ + e) * D_ + d]);
}

// g16 = f16( RATE*inv_n*stat + (1-RATE)*adp ), written in WMMA A-fragment
// order: dst = ((tm*nChunks + ck)*32 + lane)*16 + j  (nChunks = D/32)
__global__ void conv_finalize(const float* __restrict__ gstat,
                              const float* __restrict__ f,
                              const float* __restrict__ inv_n,
                              const int* __restrict__ tidx,
                              const float* __restrict__ tw,
                              _Float16* __restrict__ g16) {
    int bn = blockIdx.x;              // b*N + n  == GEMM row
    int d  = threadIdx.x;             // 128
    int n  = bn % N_;
    int b  = bn / N_;
    float s = 0.0f;
#pragma unroll
    for (int k = 0; k < TOPK_; ++k) {
        int j = tidx[(size_t)n * TOPK_ + k];
        s += tw[(size_t)n * TOPK_ + k] * f[((size_t)b * N_ + j) * D_ + d];
    }
    float val = RATE_ * inv_n[n] * gstat[(size_t)bn * D_ + d] + (1.0f - RATE_) * s;

    int tm = bn >> 4, mi = bn & 15;
    int ck = d >> 5,  r  = d & 31;
    int kh = (r & 15) >> 3;
    int jj = (r < 16) ? (r & 7) : (8 + (r & 7));
    int lane = (kh << 4) | mi;
    g16[(((size_t)tm * (D_ / 32) + ck) * 32 + lane) * 16 + jj] = (_Float16)val;
}

// ------------------------------------------------------------- WMMA GEMM
// Operands are pre-packed into fragment layout; inner loop:
//   v16h a <- 32B load, v16h b <- 32B load, v_wmma_f32_16x16x32_f16.
// Epilogues:
//   mode 0: zr = sigmoid(.)                       (Nout = 2H = 128)
//   mode 1: c = tanh(.); h = z*h+(1-z)*c; h16 pk  (Nout = H = 64)
//   mode 2: out[b,t,col,n] += .                   (Nout = C = 64, residual)
__global__ void gemm_wmma(const _Float16* __restrict__ Apack,
                          const _Float16* __restrict__ Bpack,
                          const float* __restrict__ bias,
                          int nChunks, int nTilesN, int nTiles, int mode,
                          float* __restrict__ zr, float* __restrict__ h,
                          _Float16* __restrict__ h16, float* __restrict__ out, int t) {
    int wave = blockIdx.x * (blockDim.x >> 5) + (threadIdx.x >> 5);
    if (wave >= nTiles) return;                 // uniform per wave
    int lane = threadIdx.x & 31;
    int tm = wave / nTilesN;
    int tn = wave - tm * nTilesN;

    const _Float16* ap = Apack + (((size_t)tm * nChunks) * 32 + lane) * 16;
    const _Float16* bp = Bpack + ((size_t)tn * 32 + lane) * 16;
    const size_t aStep = 32 * 16;                       // one chunk of A tile
    const size_t bStep = (size_t)nTilesN * 32 * 16;     // one chunk of all B tiles

    v8f acc = {};
    for (int c = 0; c < nChunks; ++c) {
        v16h a = *(const v16h*)ap;
        v16h b = *(const v16h*)bp;
        ap += aStep;
        bp += bStep;
        acc = __builtin_amdgcn_wmma_f32_16x16x32_f16(
                  false, a, false, b, (short)0, acc, false, false);
    }

    int mi = lane & 15;
    int kh = lane >> 4;
    int row0 = tm * 16, n0 = tn * 16;
    int ncol = n0 + mi;
    float bv = bias[ncol];
#pragma unroll
    for (int r = 0; r < 8; ++r) {
        int grow = row0 + r + 8 * kh;            // global row in [0, B*N)
        float v = acc[r] + bv;
        if (mode == 0) {
            zr[(size_t)grow * (2 * H_) + ncol] = 1.0f / (1.0f + __expf(-v));
        } else if (mode == 1) {
            float cc = tanhf(v);
            float z  = zr[(size_t)grow * (2 * H_) + ncol];
            float ho = h[(size_t)grow * H_ + ncol];
            float hn = z * ho + (1.0f - z) * cc;
            h[(size_t)grow * H_ + ncol] = hn;
            // write h16 in A-fragment order for the output-proj GEMM (K=H=64)
            int tmr = grow >> 4, mir = grow & 15;
            int ck  = ncol >> 5, rr = ncol & 31;
            int khh = (rr & 15) >> 3;
            int jj  = (rr < 16) ? (rr & 7) : (8 + (rr & 7));
            int lnp = (khh << 4) | mir;
            h16[(((size_t)tmr * (H_ / 32) + ck) * 32 + lnp) * 16 + jj] = (_Float16)hn;
        } else {
            int bb = grow / N_;
            int nn = grow - bb * N_;
            out[(((size_t)bb * T_ + t) * C_ + ncol) * N_ + nn] += v;
        }
    }
}

// ---------------------------------------------------------------------------
extern "C" void kernel_launch(void* const* d_in, const int* in_sizes, int n_in,
                              void* d_out, int out_size, void* d_ws, size_t ws_size,
                              hipStream_t stream) {
    (void)in_sizes; (void)n_in; (void)out_size; (void)ws_size;

    const float* x         = (const float*)d_in[0];
    const int*   hyper     = (const int*)d_in[1];
    const int*   node_idx  = hyper;
    const int*   edge_idx  = hyper + NNZ_;
    const float* node_embs = (const float*)d_in[2];
    const float* W_zr      = (const float*)d_in[3];
    const float* b_zr      = (const float*)d_in[4];
    const float* W_c       = (const float*)d_in[5];
    const float* b_c       = (const float*)d_in[6];
    const float* W_out     = (const float*)d_in[7];
    const float* b_out     = (const float*)d_in[8];
    float* out = (float*)d_out;

    char* ws = (char*)d_ws;
    size_t off = 0;
    auto alloc = [&](size_t bytes) -> void* {
        void* p = ws + off;
        off += (bytes + 255) & ~(size_t)255;
        return p;
    };
    float*    inv_e = (float*)alloc((size_t)NE_ * 4);
    float*    inv_n = (float*)alloc((size_t)N_ * 4);
    int*      tidx  = (int*)  alloc((size_t)L_ * N_ * TOPK_ * 4);
    float*    tw    = (float*)alloc((size_t)L_ * N_ * TOPK_ * 4);
    _Float16* Wz16  = (_Float16*)alloc((size_t)L_ * D_ * 2 * H_ * 2);
    _Float16* Wc16  = (_Float16*)alloc((size_t)L_ * D_ * H_ * 2);
    _Float16* Wo16  = (_Float16*)alloc((size_t)L_ * H_ * C_ * 2);
    float*    h     = (float*)alloc((size_t)B_ * N_ * H_ * 4);
    _Float16* h16   = (_Float16*)alloc((size_t)B_ * N_ * H_ * 2);
    float*    fbuf  = (float*)alloc((size_t)B_ * N_ * D_ * 4);
    float*    eagg  = (float*)alloc((size_t)B_ * NE_ * D_ * 4);
    float*    gstat = (float*)alloc((size_t)B_ * N_ * D_ * 4);
    _Float16* g16   = (_Float16*)alloc((size_t)B_ * N_ * D_ * 2);
    float*    zrbuf = (float*)alloc((size_t)B_ * N_ * 2 * H_ * 4);

    // out <- x (residual stream, accumulated in place per layer)
    hipMemcpyAsync(out, x, (size_t)B_ * T_ * C_ * N_ * sizeof(float),
                   hipMemcpyDeviceToDevice, stream);

    // degrees
    zero_f32<<<(NE_ + 255) / 256, 256, 0, stream>>>(inv_e, NE_);
    zero_f32<<<(N_ + 255) / 256, 256, 0, stream>>>(inv_n, N_);
    count_deg<<<(NNZ_ + 255) / 256, 256, 0, stream>>>(node_idx, edge_idx, inv_n, inv_e);
    invert_deg<<<(NE_ + 255) / 256, 256, 0, stream>>>(inv_e, NE_);
    invert_deg<<<(N_ + 255) / 256, 256, 0, stream>>>(inv_n, N_);

    // fragment-packed f16 weights (per layer)
    for (int l = 0; l < L_; ++l) {
        pack_weight<<<(D_ * 2 * H_ + 255) / 256, 256, 0, stream>>>(
            W_zr + (size_t)l * D_ * 2 * H_, Wz16 + (size_t)l * D_ * 2 * H_, D_, 2 * H_);
        pack_weight<<<(D_ * H_ + 255) / 256, 256, 0, stream>>>(
            W_c + (size_t)l * D_ * H_, Wc16 + (size_t)l * D_ * H_, D_, H_);
        pack_weight<<<(H_ * C_ + 255) / 256, 256, 0, stream>>>(
            W_out + (size_t)l * H_ * C_, Wo16 + (size_t)l * H_ * C_, H_, C_);
    }

    // adaptive top-k per layer
    for (int l = 0; l < L_; ++l) {
        topk_kernel<<<N_, 128, 0, stream>>>(node_embs + (size_t)l * N_ * DE_,
                                            tidx + (size_t)l * N_ * TOPK_,
                                            tw   + (size_t)l * N_ * TOPK_);
    }

    const long long fElems = (long long)B_ * N_ * D_;
    const int M = B_ * N_;                       // 40000 rows

    auto conv = [&](int l, int mode, int t) {
        build_f<<<(int)((fElems + 255) / 256), 256, 0, stream>>>(out, h, zrbuf, fbuf, t, mode);
        zero_f32<<<(int)(((long long)B_ * NE_ * D_ + 255) / 256), 256, 0, stream>>>(
            eagg, (long long)B_ * NE_ * D_);
        scatter_edges<<<B_ * NNZ_, D_, 0, stream>>>(fbuf, node_idx, edge_idx, eagg);
        zero_f32<<<(int)((fElems + 255) / 256), 256, 0, stream>>>(gstat, fElems);
        scatter_nodes<<<B_ * NNZ_, D_, 0, stream>>>(eagg, node_idx, edge_idx, inv_e, gstat);
        conv_finalize<<<B_ * N_, D_, 0, stream>>>(gstat, fbuf, inv_n,
                                                  tidx + (size_t)l * N_ * TOPK_,
                                                  tw   + (size_t)l * N_ * TOPK_, g16);
    };

    for (int l = 0; l < L_; ++l) {
        zero_f32<<<(int)(((long long)B_ * N_ * H_ + 255) / 256), 256, 0, stream>>>(
            h, (long long)B_ * N_ * H_);
        for (int t = 0; t < T_; ++t) {
            // zr = sigmoid(conv(concat(x_t, h)) @ Wz + bz)
            conv(l, /*mode=*/0, t);
            {
                int nTN = (2 * H_) / 16, tiles = (M / 16) * nTN;    // 20000
                gemm_wmma<<<tiles / 8, 256, 0, stream>>>(
                    g16, Wz16 + (size_t)l * D_ * 2 * H_, b_zr + (size_t)l * 2 * H_,
                    D_ / 32, nTN, tiles, 0, zrbuf, nullptr, nullptr, nullptr, t);
            }
            // c = tanh(conv(concat(x_t, r*h)) @ Wc + bc); h = z*h + (1-z)*c
            conv(l, /*mode=*/1, t);
            {
                int nTN = H_ / 16, tiles = (M / 16) * nTN;          // 10000
                gemm_wmma<<<tiles / 8, 256, 0, stream>>>(
                    g16, Wc16 + (size_t)l * D_ * H_, b_c + (size_t)l * H_,
                    D_ / 32, nTN, tiles, 1, zrbuf, h, h16, nullptr, t);
            }
            // out[b,t,:,n] += h @ Wo + bo
            {
                int nTN = C_ / 16, tiles = (M / 16) * nTN;          // 10000
                gemm_wmma<<<tiles / 8, 256, 0, stream>>>(
                    h16, Wo16 + (size_t)l * H_ * C_, b_out + (size_t)l * C_,
                    H_ / 32, nTN, tiles, 2, zrbuf, h, h16, out, t);
            }
        }
    }
}
